// MoE_87540023427082
// MI455X (gfx1250) — matrix operand order repaired and checked by reading, not compiled, not run on hardware
//
#include <hip/hip_runtime.h>

#define DIM   768
#define INTER 512
#define TOK   16384
#define NE    8
#define NE9   9

#define KT    32    // K step per WMMA stage
#define LDK   40    // padded LDS row stride (bf16 elems) -> 80B, keeps 16B alignment
#define LDKB  80    // row stride in bytes
#define MTM   128   // M tile per workgroup
#define NTN   256   // N tile per workgroup

typedef __bf16 v16bf __attribute__((ext_vector_type(16)));
typedef __bf16 v8bf  __attribute__((ext_vector_type(8)));
typedef float  v8f   __attribute__((ext_vector_type(8)));
typedef unsigned short bf16_raw;

// ---------------------------------------------------------------------------
// CDNA5 async DMA: copy 16 bytes global -> LDS, tracked by ASYNCcnt.
// lds_off = wave-relative LDS byte address (low 32 bits of flat pointer).
// ---------------------------------------------------------------------------
__device__ __forceinline__ void async_copy16(unsigned lds_off, const __bf16* g) {
    asm volatile("global_load_async_to_lds_b128 %0, %1, off"
                 :: "v"(lds_off), "v"(g) : "memory");
}
__device__ __forceinline__ unsigned lds_addr(const void* p) {
    return (unsigned)(unsigned long long)p;
}

// Assemble a v16bf WMMA operand from two 16-byte LDS chunks (each 16B aligned).
__device__ __forceinline__ v16bf ld_op(const __bf16* p0, const __bf16* p1) {
    v8bf lo = *(const v8bf*)p0;
    v8bf hi = *(const v8bf*)p1;
    v16bf r;
#pragma unroll
    for (int i = 0; i < 8; ++i) { r[i] = lo[i]; r[i + 8] = hi[i]; }
    return r;
}

// ---------------------------------------------------------------------------
// One-shot f32 -> bf16 conversion (8 elems / thread, vectorized)
// ---------------------------------------------------------------------------
__global__ __launch_bounds__(256)
void cvt_bf16_kernel(const float* __restrict__ src, bf16_raw* __restrict__ dst, int n8) {
    int i = blockIdx.x * 256 + threadIdx.x;
    if (i >= n8) return;
    const float4* s = (const float4*)src + (size_t)i * 2;
    float4 a = s[0], b = s[1];
    v8bf o;
    o[0] = (__bf16)a.x; o[1] = (__bf16)a.y; o[2] = (__bf16)a.z; o[3] = (__bf16)a.w;
    o[4] = (__bf16)b.x; o[5] = (__bf16)b.y; o[6] = (__bf16)b.z; o[7] = (__bf16)b.w;
    ((v8bf*)dst)[i] = o;
}

// ---------------------------------------------------------------------------
// Gating: wave-per-token sigmoid scores + grouped top-k (ties -> lowest index).
// cw[t, 0..7] = normalized routed weights (0 if unselected); cw[t, 8] = 1.0
// ---------------------------------------------------------------------------
__global__ __launch_bounds__(256)
void moe_gate_kernel(const float* __restrict__ x,
                     const float* __restrict__ gw,
                     const float* __restrict__ gb,
                     float* __restrict__ cw) {
    __shared__ float sgw[NE * DIM];
    __shared__ float sgb[NE];
    for (int i = threadIdx.x; i < NE * DIM; i += 256) sgw[i] = gw[i];
    if (threadIdx.x < NE) sgb[threadIdx.x] = gb[threadIdx.x];
    __syncthreads();

    const int wave = threadIdx.x >> 5;
    const int lane = threadIdx.x & 31;
    const int tok  = blockIdx.x * 8 + wave;

    float p[NE];
#pragma unroll
    for (int e = 0; e < NE; ++e) p[e] = 0.f;

    for (int d = lane; d < DIM; d += 32) {
        float xv = x[tok * DIM + d];
#pragma unroll
        for (int e = 0; e < NE; ++e) p[e] += xv * sgw[e * DIM + d];
    }
#pragma unroll
    for (int e = 0; e < NE; ++e)
        for (int off = 16; off > 0; off >>= 1) p[e] += __shfl_xor(p[e], off, 32);

    if (lane == 0) {
        float scores[NE], s[NE];
#pragma unroll
        for (int e = 0; e < NE; ++e) {
            scores[e] = 1.f / (1.f + __expf(-p[e]));
            s[e] = scores[e] + sgb[e];
        }
        float gs[4];
#pragma unroll
        for (int g = 0; g < 4; ++g) gs[g] = s[2 * g] + s[2 * g + 1];
        int g1 = 0;
        for (int g = 1; g < 4; ++g) if (gs[g] > gs[g1]) g1 = g;
        int g2 = -1;
        for (int g = 0; g < 4; ++g) {
            if (g == g1) continue;
            if (g2 < 0 || gs[g] > gs[g2]) g2 = g;
        }
        int i1 = -1;
        for (int e = 0; e < NE; ++e) {
            int g = e >> 1;
            if (g != g1 && g != g2) continue;
            if (i1 < 0 || s[e] > s[i1]) i1 = e;
        }
        int i2 = -1;
        for (int e = 0; e < NE; ++e) {
            int g = e >> 1;
            if ((g != g1 && g != g2) || e == i1) continue;
            if (i2 < 0 || s[e] > s[i2]) i2 = e;
        }
        float wa = scores[i1], wb = scores[i2];
        float inv = 1.f / (wa + wb + 1e-6f);
        for (int e = 0; e < NE9; ++e) {
            float v = 0.f;
            if (e == i1) v = wa * inv;
            else if (e == i2) v = wb * inv;
            else if (e == 8) v = 1.f;
            cw[tok * NE9 + e] = v;
        }
    }
}

// ---------------------------------------------------------------------------
// Stage 1 (per expert): H[t,i] = cw[t,e] * silu(x@w1^T) * (x@w3^T)  (bf16 out)
// WG tile 128x256, 8 waves (2M x 4N) of 64x64; async double-buffered staging.
// ---------------------------------------------------------------------------
__global__ __launch_bounds__(256)
void moe_g13_kernel(const bf16_raw* __restrict__ xbr,
                    const bf16_raw* __restrict__ w1r,
                    const bf16_raw* __restrict__ w3r,
                    const float* __restrict__ cw,
                    bf16_raw* __restrict__ Hraw,
                    int expert) {
    __shared__ __bf16 sA[2][MTM * LDK];
    __shared__ __bf16 sB1[2][NTN * LDK];
    __shared__ __bf16 sB3[2][NTN * LDK];
    const __bf16* xb = (const __bf16*)xbr;
    const __bf16* w1 = (const __bf16*)w1r;
    const __bf16* w3 = (const __bf16*)w3r;
    __bf16* H = (__bf16*)Hraw;

    const int tid   = threadIdx.x;
    const int wave  = tid >> 5;
    const int lane  = tid & 31;
    const int l     = lane & 15;
    const int half  = lane >> 4;
    const int waveM = wave >> 2;   // 0..1 -> M offset *64
    const int waveN = wave & 3;    // 0..3 -> N offset *64
    const int m0 = blockIdx.x * MTM;
    const int n0 = blockIdx.y * NTN;

    // staging: 16B chunks; each tile row of 32 bf16 = 4 chunks
    const int rr = tid >> 2;            // rows 0..63 (+64*i)
    const int ce = (tid & 3) * 8;       // element col offset
    const unsigned cb = (tid & 3) * 16; // byte col offset in LDS row

    v8f acc1[4][4], acc3[4][4];
#pragma unroll
    for (int t = 0; t < 4; ++t)
#pragma unroll
        for (int n = 0; n < 4; ++n)
#pragma unroll
            for (int i = 0; i < 8; ++i) { acc1[t][n][i] = 0.f; acc3[t][n][i] = 0.f; }

    auto issue = [&](int kk, int b) {
        const __bf16* ga = xb + (size_t)m0 * DIM + kk + ce;
        const __bf16* g1 = w1 + (size_t)n0 * DIM + kk + ce;
        const __bf16* g3 = w3 + (size_t)n0 * DIM + kk + ce;
        unsigned la  = lds_addr(&sA[b][0])  + cb;
        unsigned lb1 = lds_addr(&sB1[b][0]) + cb;
        unsigned lb3 = lds_addr(&sB3[b][0]) + cb;
#pragma unroll
        for (int i = 0; i < 2; ++i)  // A: 128 rows
            async_copy16(la + (rr + 64 * i) * LDKB, ga + (size_t)(rr + 64 * i) * DIM);
#pragma unroll
        for (int i = 0; i < 4; ++i) {  // B1/B3: 256 rows each
            async_copy16(lb1 + (rr + 64 * i) * LDKB, g1 + (size_t)(rr + 64 * i) * DIM);
            async_copy16(lb3 + (rr + 64 * i) * LDKB, g3 + (size_t)(rr + 64 * i) * DIM);
        }
    };

    const int NKI = DIM / KT;   // 24
    issue(0, 0);
    for (int k = 0; k < NKI; ++k) {
        const int b = k & 1;
        if (k + 1 < NKI) {
            issue((k + 1) * KT, (k + 1) & 1);
            asm volatile("s_wait_asynccnt 0xa" ::: "memory"); // tile k landed
        } else {
            asm volatile("s_wait_asynccnt 0x0" ::: "memory");
        }
        __syncthreads();

        v16bf a[4];
#pragma unroll
        for (int t = 0; t < 4; ++t) {
            const int ra = (waveM * 64 + t * 16 + l) * LDK;
            // A layout: lane<16 -> K{0..15}; lane>=16 -> K{16..31}
            a[t] = ld_op(&sA[b][ra + half * 16], &sA[b][ra + half * 16 + 8]);
        }
#pragma unroll
        for (int n = 0; n < 4; ++n) {
            const int rb = (waveN * 64 + n * 16 + l) * LDK + half * 16;
            // B layout: lane<16 -> col n, K{0..15}; lane>=16 -> K{16..31}
            v16bf b1v = ld_op(&sB1[b][rb], &sB1[b][rb + 8]);
            v16bf b3v = ld_op(&sB3[b][rb], &sB3[b][rb + 8]);
#pragma unroll
            for (int t = 0; t < 4; ++t) {
                acc1[t][n] = __builtin_amdgcn_wmma_f32_16x16x32_bf16(
                    false, a[t], false, b1v, (short)0, acc1[t][n], false, false);
                acc3[t][n] = __builtin_amdgcn_wmma_f32_16x16x32_bf16(
                    false, a[t], false, b3v, (short)0, acc3[t][n], false, false);
            }
        }
        __syncthreads();  // all waves done reading buf b before it is re-filled
    }

    // Epilogue: h = silu(h1) * h3 * cw[row, e]  ->  bf16 H[row, col]
#pragma unroll
    for (int t = 0; t < 4; ++t) {
#pragma unroll
        for (int v = 0; v < 8; ++v) {
            const int row = m0 + waveM * 64 + t * 16 + v + half * 8;
            const float c = cw[row * NE9 + expert];
#pragma unroll
            for (int n = 0; n < 4; ++n) {
                const float a1 = acc1[t][n][v];
                const float a3 = acc3[t][n][v];
                const float sl = a1 / (1.f + __expf(-a1));
                const float hv = sl * a3 * c;
                const int col = n0 + waveN * 64 + n * 16 + l;
                H[(size_t)row * INTER + col] = (__bf16)hv;
            }
        }
    }
}

// ---------------------------------------------------------------------------
// Stage 2 (per expert): out[T,768] (+)= H[T,512] @ w2[768,512]^T   (bf16 ops)
// WG tile 128x256, 8 waves (2M x 4N) of 64x64.
// ---------------------------------------------------------------------------
__global__ __launch_bounds__(256)
void moe_g2_kernel(const bf16_raw* __restrict__ Hraw,
                   const bf16_raw* __restrict__ w2r,
                   float* __restrict__ out,
                   int accumulate) {
    __shared__ __bf16 sA[2][MTM * LDK];
    __shared__ __bf16 sB[2][NTN * LDK];
    const __bf16* H  = (const __bf16*)Hraw;
    const __bf16* w2 = (const __bf16*)w2r;

    const int tid   = threadIdx.x;
    const int wave  = tid >> 5;
    const int lane  = tid & 31;
    const int l     = lane & 15;
    const int half  = lane >> 4;
    const int waveM = wave >> 2;
    const int waveN = wave & 3;
    const int m0 = blockIdx.x * MTM;
    const int n0 = blockIdx.y * NTN;

    const int rr = tid >> 2;
    const int ce = (tid & 3) * 8;
    const unsigned cb = (tid & 3) * 16;

    v8f acc[4][4];
#pragma unroll
    for (int t = 0; t < 4; ++t)
#pragma unroll
        for (int n = 0; n < 4; ++n)
#pragma unroll
            for (int i = 0; i < 8; ++i) acc[t][n][i] = 0.f;

    auto issue = [&](int kk, int b) {
        const __bf16* ga = H  + (size_t)m0 * INTER + kk + ce;
        const __bf16* gb = w2 + (size_t)n0 * INTER + kk + ce;
        unsigned la = lds_addr(&sA[b][0]) + cb;
        unsigned lb = lds_addr(&sB[b][0]) + cb;
#pragma unroll
        for (int i = 0; i < 2; ++i)
            async_copy16(la + (rr + 64 * i) * LDKB, ga + (size_t)(rr + 64 * i) * INTER);
#pragma unroll
        for (int i = 0; i < 4; ++i)
            async_copy16(lb + (rr + 64 * i) * LDKB, gb + (size_t)(rr + 64 * i) * INTER);
    };

    const int NKI = INTER / KT;  // 16
    issue(0, 0);
    for (int k = 0; k < NKI; ++k) {
        const int b = k & 1;
        if (k + 1 < NKI) {
            issue((k + 1) * KT, (k + 1) & 1);
            asm volatile("s_wait_asynccnt 0x6" ::: "memory");
        } else {
            asm volatile("s_wait_asynccnt 0x0" ::: "memory");
        }
        __syncthreads();

        v16bf a[4];
#pragma unroll
        for (int t = 0; t < 4; ++t) {
            const int ra = (waveM * 64 + t * 16 + l) * LDK;
            a[t] = ld_op(&sA[b][ra + half * 16], &sA[b][ra + half * 16 + 8]);
        }
#pragma unroll
        for (int n = 0; n < 4; ++n) {
            const int rb = (waveN * 64 + n * 16 + l) * LDK + half * 16;
            v16bf bv = ld_op(&sB[b][rb], &sB[b][rb + 8]);
#pragma unroll
            for (int t = 0; t < 4; ++t) {
                acc[t][n] = __builtin_amdgcn_wmma_f32_16x16x32_bf16(
                    false, a[t], false, bv, (short)0, acc[t][n], false, false);
            }
        }
        __syncthreads();
    }

#pragma unroll
    for (int t = 0; t < 4; ++t) {
#pragma unroll
        for (int v = 0; v < 8; ++v) {
            const int row = m0 + waveM * 64 + t * 16 + v + half * 8;
#pragma unroll
            for (int n = 0; n < 4; ++n) {
                const int col = n0 + waveN * 64 + n * 16 + l;
                const size_t idx = (size_t)row * DIM + col;
                const float val = acc[t][n][v];
                out[idx] = accumulate ? (out[idx] + val) : val;
            }
        }
    }
}

// ---------------------------------------------------------------------------
extern "C" void kernel_launch(void* const* d_in, const int* in_sizes, int n_in,
                              void* d_out, int out_size, void* d_ws, size_t ws_size,
                              hipStream_t stream) {
    const float* x   = (const float*)d_in[0];
    const float* gw  = (const float*)d_in[1];
    const float* gb  = (const float*)d_in[2];
    const float* w1  = (const float*)d_in[3];
    const float* w2  = (const float*)d_in[4];
    const float* w3  = (const float*)d_in[5];
    const float* sw1 = (const float*)d_in[6];
    const float* sw2 = (const float*)d_in[7];
    const float* sw3 = (const float*)d_in[8];
    float* out = (float*)d_out;

    char* ws = (char*)d_ws;
    size_t off = 0;
    float* cw = (float*)(ws + off);         off += (size_t)TOK * NE9 * 4;       // 576 KB
    bf16_raw* Hb   = (bf16_raw*)(ws + off); off += (size_t)TOK * INTER * 2;     // 16.8 MB
    bf16_raw* xb   = (bf16_raw*)(ws + off); off += (size_t)TOK * DIM * 2;       // 25.2 MB
    bf16_raw* w1b  = (bf16_raw*)(ws + off); off += (size_t)NE * INTER * DIM * 2;
    bf16_raw* w3b  = (bf16_raw*)(ws + off); off += (size_t)NE * INTER * DIM * 2;
    bf16_raw* w2b  = (bf16_raw*)(ws + off); off += (size_t)NE * DIM * INTER * 2;
    bf16_raw* sw1b = (bf16_raw*)(ws + off); off += (size_t)INTER * DIM * 2;
    bf16_raw* sw3b = (bf16_raw*)(ws + off); off += (size_t)INTER * DIM * 2;
    bf16_raw* sw2b = (bf16_raw*)(ws + off); off += (size_t)DIM * INTER * 2;

    moe_gate_kernel<<<TOK / 8, 256, 0, stream>>>(x, gw, gb, cw);

    const int nx  = TOK * DIM / 8;
    const int nw  = NE * INTER * DIM / 8;
    const int nsw = INTER * DIM / 8;
    cvt_bf16_kernel<<<(nx  + 255) / 256, 256, 0, stream>>>(x,   xb,   nx);
    cvt_bf16_kernel<<<(nw  + 255) / 256, 256, 0, stream>>>(w1,  w1b,  nw);
    cvt_bf16_kernel<<<(nw  + 255) / 256, 256, 0, stream>>>(w3,  w3b,  nw);
    cvt_bf16_kernel<<<(nw  + 255) / 256, 256, 0, stream>>>(w2,  w2b,  nw);
    cvt_bf16_kernel<<<(nsw + 255) / 256, 256, 0, stream>>>(sw1, sw1b, nsw);
    cvt_bf16_kernel<<<(nsw + 255) / 256, 256, 0, stream>>>(sw3, sw3b, nsw);
    cvt_bf16_kernel<<<(nsw + 255) / 256, 256, 0, stream>>>(sw2, sw2b, nsw);

    for (int e = 0; e < NE9; ++e) {
        const bf16_raw* pw1 = (e < NE) ? (w1b + (size_t)e * INTER * DIM) : sw1b;
        const bf16_raw* pw3 = (e < NE) ? (w3b + (size_t)e * INTER * DIM) : sw3b;
        const bf16_raw* pw2 = (e < NE) ? (w2b + (size_t)e * DIM * INTER) : sw2b;
        moe_g13_kernel<<<dim3(TOK / MTM, INTER / NTN), 256, 0, stream>>>(
            xb, pw1, pw3, cw, Hb, e);
        moe_g2_kernel<<<dim3(TOK / MTM, DIM / NTN), 256, 0, stream>>>(
            Hb, pw2, out, e != 0);
    }
}
